// GRU_CNN_Attention_3332894622213
// MI455X (gfx1250) — matrix operand order repaired and checked by reading, not compiled, not run on hardware
//
#include <hip/hip_runtime.h>

// ---------------- problem constants ----------------
#define BDOC   32
#define SDIM   50
#define SL     100          // L
#define EDIM   200          // E
#define GDIM   128          // G
#define N3G    384          // 3*G
#define D2DIM  256          // 2*G
#define CDIM   100          // C
#define CD3    300          // 3*C
#define NSENT  1600         // B*S
#define KXPAD  224          // E padded to multiple of 32
#define MAXR   20

typedef __bf16 bf16;
typedef __attribute__((ext_vector_type(8)))  __bf16 v8bf;
typedef __attribute__((ext_vector_type(16))) __bf16 v16bf;
typedef __attribute__((ext_vector_type(8)))  float  v8f;
typedef __attribute__((ext_vector_type(4)))  unsigned int v4u;
typedef __attribute__((ext_vector_type(8)))  int v8i;
typedef __attribute__((ext_vector_type(4)))  int v4i;

__device__ inline v16bf cat16(v8bf lo, v8bf hi) {
    v16bf r;
#pragma unroll
    for (int i = 0; i < 8; ++i) { r[i] = lo[i]; r[i + 8] = hi[i]; }
    return r;
}

// A fragment (16x32 bf16, M x K). lane = m (0..15), halves split K per ISA layout:
// lanes 0-15: K = kbase+0..7 and kbase+16..23 ; lanes 16-31: +8.
__device__ inline v16bf load_A(const bf16* base, int stride, int kbase) {
    int l  = threadIdx.x & 31;
    int m  = l & 15;
    int h8 = (l >> 4) * 8;
    const bf16* p = base + m * stride + kbase + h8;
    v8bf lo = *(const v8bf*)p;
    v8bf hi = *(const v8bf*)(p + 16);
    return cat16(lo, hi);
}

// B fragment (32x16 bf16, K x N) from n-major weight W[n][k] (stride = Kpad).
// lane = n (0..15); lanes 0-15 hold K = kbase+0..7 and kbase+16..23 ; lanes 16-31: +8.
__device__ inline v16bf load_B(const bf16* base, int stride, int kbase) {
    int l   = threadIdx.x & 31;
    int n   = l & 15;
    int h16 = (l >> 4) * 8;
    const bf16* p = base + n * stride + kbase + h16;
    v8bf lo = *(const v8bf*)p;
    v8bf hi = *(const v8bf*)(p + 16);
    return cat16(lo, hi);
}

__device__ inline v8f wmma_bf16(v16bf a, v16bf b, v8f c) {
    return __builtin_amdgcn_wmma_f32_16x16x32_bf16(false, a, false, b, (short)0, c, false, false);
}

// TDM: contiguous copy of nelem bf16 elements global -> LDS (one 1-row 2D tile).
// D# per CDNA5 ISA ch.8: group0 {count=1, lds_addr, global_addr[56:0], type=2},
// group1 {data_size=2B, tensor_dim0=tile_dim0=nelem, tensor_dim1=tile_dim1=1, stride0=nelem}.
// This toolchain exposes the 6-arg builtin: groups 0..3 + trailing 8-dword group + cpol.
__device__ inline void tdm_load_to_lds(const void* gsrc, void* lds_dst, unsigned int nelem) {
    unsigned long long ga = (unsigned long long)(uintptr_t)gsrc;
    unsigned int       la = (unsigned int)(uintptr_t)lds_dst;   // low 32 bits = LDS offset
    v4u g0 = { 1u,                                              // count=1, user mode
               la,
               (unsigned int)ga,
               (unsigned int)((ga >> 32) & 0x1FFFFFFu) | (2u << 30) };  // addr[56:32] | type=2
    v8i g1 = { (int)(1u << 16),                                 // data_size = 1 -> 2 bytes
               (int)((nelem & 0xFFFFu) << 16),                  // tensor_dim0 lo16
               (int)(((nelem >> 16) & 0xFFFFu) | (1u << 16)),   // tensor_dim0 hi16 | tensor_dim1 lo16 (=1)
               (int)((nelem & 0xFFFFu) << 16),                  // tensor_dim1 hi16 (=0) | tile_dim0
               1,                                               // tile_dim1 = 1, tile_dim2 = 0
               (int)nelem,                                      // tensor_dim0_stride lo32
               0, 0 };
    v4i g2 = { 0, 0, 0, 0 };
    v4i g3 = { 0, 0, 0, 0 };
    v8i g4 = { 0, 0, 0, 0, 0, 0, 0, 0 };
    __builtin_amdgcn_tensor_load_to_lds(g0, g1, g2, g3, g4, 0);
}

// ---------------- weight preparation ----------------
__global__ void k_convert_pad(const float* __restrict__ src, bf16* __restrict__ dst,
                              int rows, int k, int kp) {
    int i = blockIdx.x * blockDim.x + threadIdx.x;
    if (i >= rows * kp) return;
    int r = i / kp, c = i % kp;
    dst[i] = (c < k) ? (bf16)src[r * k + c] : (bf16)0.0f;
}

// conv weight (K,100) HIO-flat -> transposed n-major (Np=112, K) with n>=100 zero
__global__ void k_conv_t(const float* __restrict__ src, bf16* __restrict__ dst,
                         int K, int Np) {
    int i = blockIdx.x * blockDim.x + threadIdx.x;
    if (i >= Np * K) return;
    int n = i / K, q = i % K;
    dst[i] = (n < CDIM) ? (bf16)src[q * CDIM + n] : (bf16)0.0f;
}

// ---------------- BiGRU: fused embed-gather + x-proj + recurrence ----------------
// grid (NSENT/16, 2). Each WG owns 16 batch rows through all 100 steps (recurrence
// is independent per row -> no inter-WG sync needed).
__global__ __launch_bounds__(256) void k_gru(
    const int* __restrict__ x, const int* __restrict__ pm, const float* __restrict__ embed,
    const bf16* __restrict__ WihT_f, const bf16* __restrict__ WhhT_f,
    const float* __restrict__ bih_f, const float* __restrict__ bhh_f,
    const bf16* __restrict__ WihT_b, const bf16* __restrict__ WhhT_b,
    const float* __restrict__ bih_b, const float* __restrict__ bhh_b,
    float* __restrict__ wo, bf16* __restrict__ wo_bf)
{
    __shared__ __align__(16) bf16 Ax[16][KXPAD];   // bf16 embeddings (K padded)
    __shared__ __align__(16) bf16 Ah[16][GDIM];    // bf16 hidden state
    __shared__ float Gx[16][N3G];
    __shared__ float Gh[16][N3G];
    __shared__ float hS[16][GDIM];                 // fp32 hidden carry
    __shared__ int   toks[16];

    const int tid = threadIdx.x, wave = tid >> 5, lane = tid & 31;
    const int chunk = blockIdx.x, dir = blockIdx.y;
    const bf16*  WihT = dir ? WihT_b : WihT_f;
    const bf16*  WhhT = dir ? WhhT_b : WhhT_f;
    const float* bih  = dir ? bih_b  : bih_f;
    const float* bhh  = dir ? bhh_b  : bhh_f;

    for (int i = tid; i < 16 * KXPAD; i += 256) (&Ax[0][0])[i] = (bf16)0.0f;
    for (int i = tid; i < 16 * GDIM; i += 256) { (&hS[0][0])[i] = 0.0f; (&Ah[0][0])[i] = (bf16)0.0f; }
    __syncthreads();

    for (int s = 0; s < SL; ++s) {
        const int t = dir ? (SL - 1 - s) : s;
        if (tid < 16) toks[tid] = x[(chunk * 16 + tid) * SL + t];
        __syncthreads();
        {   // gather + convert embedding rows (16 rows x 200)
            int m = tid >> 4, k0 = tid & 15;
            const float* er = embed + (size_t)toks[m] * EDIM;
            for (int k = k0; k < EDIM; k += 16) Ax[m][k] = (bf16)er[k];
        }
        __syncthreads();

        // Gx = emb @ Wih^T ; Gh = h @ Whh^T   (24 N-tiles over 8 waves)
        for (int q = 0; q < 3; ++q) {
            const int nt = wave * 3 + q;
            v8f ax = {};
#pragma unroll
            for (int kt = 0; kt < KXPAD / 32; ++kt)
                ax = wmma_bf16(load_A(&Ax[0][0], KXPAD, kt * 32),
                               load_B(WihT + nt * 16 * KXPAD, KXPAD, kt * 32), ax);
            v8f ah = {};
#pragma unroll
            for (int kt = 0; kt < GDIM / 32; ++kt)
                ah = wmma_bf16(load_A(&Ah[0][0], GDIM, kt * 32),
                               load_B(WhhT + nt * 16 * GDIM, GDIM, kt * 32), ah);
            const int n = nt * 16 + (lane & 15), mh = (lane >> 4) * 8;
#pragma unroll
            for (int j = 0; j < 8; ++j) { Gx[mh + j][n] = ax[j]; Gh[mh + j][n] = ah[j]; }
        }
        __syncthreads();

        // gates
        for (int i = tid; i < 16 * GDIM; i += 256) {
            const int m = i >> 7, c = i & 127;
            float xr = Gx[m][c]       + bih[c];
            float xz = Gx[m][c + 128] + bih[c + 128];
            float xn = Gx[m][c + 256] + bih[c + 256];
            float hr = Gh[m][c]       + bhh[c];
            float hz = Gh[m][c + 128] + bhh[c + 128];
            float hn = Gh[m][c + 256] + bhh[c + 256];
            float r  = 1.0f / (1.0f + __expf(-(xr + hr)));
            float z  = 1.0f / (1.0f + __expf(-(xz + hz)));
            float nn = tanhf(xn + r * hn);
            float hp = hS[m][c];
            float hw = (1.0f - z) * nn + z * hp;
            const int sent = chunk * 16 + m;
            const int mk   = pm[sent * SL + t];
            float hk = mk ? hw : hp;
            float ov = mk ? hw : 0.0f;
            hS[m][c] = hk;
            Ah[m][c] = (bf16)hk;
            const size_t oidx = ((size_t)sent * SL + t) * D2DIM + dir * GDIM + c;
            wo[oidx]    = ov;
            wo_bf[oidx] = (bf16)ov;
        }
        __syncthreads();
    }
}

// ---------------- per-sentence: conv(k=2,3,4)+maxpool and word attention ----------------
__global__ __launch_bounds__(256) void k_sentence(
    const float* __restrict__ wo, const bf16* __restrict__ wo_bf, const int* __restrict__ pm,
    const bf16* __restrict__ cT0, const bf16* __restrict__ cT1, const bf16* __restrict__ cT2,
    const float* __restrict__ cb0, const float* __restrict__ cb1, const float* __restrict__ cb2,
    const bf16* __restrict__ wfcT, const float* __restrict__ wfcb, const float* __restrict__ wq,
    float* __restrict__ sv, float* __restrict__ pair)
{
    __shared__ __align__(16) bf16 woS[116 * D2DIM];   // rows 100..115 zero pad
    __shared__ float partials[7][112];
    __shared__ float scoreS[112];
    __shared__ float wS[112];

    const int tid = threadIdx.x, wave = tid >> 5, lane = tid & 31;
    const int sent = blockIdx.x;
    const float* wrow = wo + (size_t)sent * SL * D2DIM;

    // DMA the bf16 wo tile into LDS with the Tensor Data Mover while the other
    // waves zero the pad rows; wave 0 waits on TENSORcnt, barrier publishes LDS.
    if (wave == 0)
        tdm_load_to_lds(wo_bf + (size_t)sent * SL * D2DIM, woS, SL * D2DIM);
    for (int i = tid; i < 16 * D2DIM; i += 256) woS[SL * D2DIM + i] = (bf16)0.0f;
    if (tid < 112) scoreS[tid] = 0.0f;
    if (wave == 0) __builtin_amdgcn_s_wait_tensorcnt(0);
    __syncthreads();

    const bf16*  cT[3] = {cT0, cT1, cT2};
    const float* cb[3] = {cb0, cb1, cb2};

    // ---- convolutions as im2col GEMM (A rows contiguous due to window overlap) ----
    for (int ks = 0; ks < 3; ++ks) {
        const int Kc = (ks + 2) * D2DIM;     // 512 / 768 / 1024
        const int Pv = SL - (ks + 1);        // 99 / 98 / 97 valid positions
        if (wave < 7) {
            for (int nt = 0; nt < 7; ++nt) {
                v8f acc = {};
                for (int kt = 0; kt < Kc / 32; ++kt)
                    acc = wmma_bf16(load_A(woS + wave * 16 * D2DIM, D2DIM, kt * 32),
                                    load_B(cT[ks] + nt * 16 * Kc, Kc, kt * 32), acc);
                const int mh = (lane >> 4) * 8, p0 = wave * 16 + mh;
                float mx = -1e30f;
#pragma unroll
                for (int j = 0; j < 8; ++j) if (p0 + j < Pv) mx = fmaxf(mx, acc[j]);
                mx = fmaxf(mx, __shfl_xor(mx, 16, 32));
                if ((lane >> 4) == 0) partials[wave][nt * 16 + (lane & 15)] = mx;
            }
        }
        __syncthreads();
        if (tid < CDIM) {
            float mx = -1e30f;
            for (int w = 0; w < 7; ++w) mx = fmaxf(mx, partials[w][tid]);
            sv[sent * CD3 + ks * CDIM + tid] = mx + cb[ks][tid];
        }
        __syncthreads();
    }

    // ---- word attention: score[p] = sum_n tanh((wo @ Wfc^T)[p,n] + b[n]) * q[n] ----
    if (wave < 7) {
        for (int nt = 0; nt < 16; ++nt) {
            v8f acc = {};
#pragma unroll
            for (int kt = 0; kt < 8; ++kt)
                acc = wmma_bf16(load_A(woS + wave * 16 * D2DIM, D2DIM, kt * 32),
                                load_B(wfcT + nt * 16 * D2DIM, D2DIM, kt * 32), acc);
            const int n = nt * 16 + (lane & 15), mh = (lane >> 4) * 8;
            const float qv = wq[n], fb = wfcb[n];
            float sj[8];
#pragma unroll
            for (int j = 0; j < 8; ++j) sj[j] = tanhf(acc[j] + fb) * qv;
            for (int d = 1; d < 16; d <<= 1)
#pragma unroll
                for (int j = 0; j < 8; ++j) sj[j] += __shfl_xor(sj[j], d, 32);
            if ((lane & 15) == 0)
#pragma unroll
                for (int j = 0; j < 8; ++j) scoreS[wave * 16 + mh + j] += sj[j];
        }
    }
    __syncthreads();

    if (tid == 0) {   // softmax over all L, mask, renormalize
        float mx = -1e30f;
        for (int p = 0; p < SL; ++p) mx = fmaxf(mx, scoreS[p]);
        float sum = 0.0f;
        for (int p = 0; p < SL; ++p) { float e = __expf(scoreS[p] - mx); wS[p] = e; sum += e; }
        float msum = 0.0f;
        for (int p = 0; p < SL; ++p) {
            float v = wS[p] / sum * (float)pm[sent * SL + p];
            wS[p] = v; msum += v;
        }
        float inv = 1.0f / (msum + 0.0001f);
        for (int p = 0; p < SL; ++p) wS[p] *= inv;
    }
    __syncthreads();

    {   // pair[c] = sum_p w[p] * wo[p][c]   (fp32 from global for accuracy)
        float a = 0.0f;
        for (int p = 0; p < SL; ++p) a += wS[p] * wrow[p * D2DIM + tid];
        pair[(size_t)sent * D2DIM + tid] = a;
    }
}

// ---------------- sentence attention score ----------------
__global__ __launch_bounds__(256) void k_sentscore(
    const float* __restrict__ sv, const float* __restrict__ W, const float* __restrict__ b,
    const float* __restrict__ q, float* __restrict__ sscore)
{
    __shared__ float svS[CD3];
    __shared__ float red[256];
    const int tid = threadIdx.x, sent = blockIdx.x;
    for (int i = tid; i < CD3; i += 256) svS[i] = sv[sent * CD3 + i];
    __syncthreads();
    float part = 0.0f;
    for (int n = tid; n < CD3; n += 256) {
        float d = b[n];
        const float* wr = W + n * CD3;
        for (int k = 0; k < CD3; ++k) d += svS[k] * wr[k];
        part += tanhf(d) * q[n];
    }
    red[tid] = part; __syncthreads();
    for (int s2 = 128; s2 > 0; s2 >>= 1) { if (tid < s2) red[tid] += red[tid + s2]; __syncthreads(); }
    if (tid == 0) sscore[sent] = red[0];
}

// ---------------- per-document: doc vector, commonality, MLP, softmax ----------------
__global__ __launch_bounds__(256) void k_doc(
    const float* __restrict__ sv, const float* __restrict__ pair,
    const float* __restrict__ sscore, const int* __restrict__ tmask,
    const float* __restrict__ W1, const float* __restrict__ b1,
    const float* __restrict__ W2, const float* __restrict__ b2, float* __restrict__ out)
{
    __shared__ float swS[SDIM], selS[SDIM];
    __shared__ float allv[CD3 + D2DIM];
    __shared__ float h1[GDIM];
    __shared__ float lS;
    const int tid = threadIdx.x, doc = blockIdx.x;
    if (tid == 0) {
        float mx = -1e30f;
        for (int s = 0; s < SDIM; ++s) mx = fmaxf(mx, sscore[doc * SDIM + s]);
        float sum = 0.0f;
        for (int s = 0; s < SDIM; ++s) { float e = __expf(sscore[doc * SDIM + s] - mx); swS[s] = e; sum += e; }
        float Lc = 0.0f;
        for (int s = 0; s < SDIM; ++s) Lc += (float)tmask[doc * SDIM + s];
        float msum = 0.0f;
        for (int s = 0; s < SDIM; ++s) {
            float v = swS[s] / sum * (float)tmask[doc * SDIM + s];
            swS[s] = v; msum += v;
        }
        float inv = 1.0f / (msum + 0.0001f);
        for (int s = 0; s < SDIM; ++s) swS[s] *= inv;
        float cum = 0.0f;
        for (int s = 0; s < SDIM; ++s) {
            float tv = (float)tmask[doc * SDIM + s];
            cum += tv;
            selS[s] = (tv > 0.0f && cum > Lc - (float)MAXR) ? 1.0f : 0.0f;
        }
        lS = fminf(Lc, (float)MAXR);
    }
    __syncthreads();
    if (tid < CD3) {
        float a = 0.0f;
        for (int s = 0; s < SDIM; ++s) a += swS[s] * sv[((size_t)doc * SDIM + s) * CD3 + tid];
        allv[tid] = a;
    }
    if (tid < D2DIM) {
        float s1 = 0.0f, s2 = 0.0f;
        for (int s = 0; s < SDIM; ++s) {
            float p = pair[((size_t)doc * SDIM + s) * D2DIM + tid];
            float sl = selS[s];
            s1 += sl * p; s2 += sl * p * p;
        }
        float sp = (s1 * s1 - s2) / (2.0f * lS);
        allv[CD3 + tid] = sp >= 0.0f ? sp : 0.01f * sp;
    }
    __syncthreads();
    if (tid < GDIM) {
        float a = b1[tid];
        for (int i = 0; i < CD3 + D2DIM; ++i) a += allv[i] * W1[i * GDIM + tid];
        h1[tid] = fmaxf(a, 0.0f);
    }
    __syncthreads();
    if (tid == 0) {
        float l0 = b2[0], l1 = b2[1];
        for (int g = 0; g < GDIM; ++g) { l0 += h1[g] * W2[g * 2]; l1 += h1[g] * W2[g * 2 + 1]; }
        float mx = fmaxf(l0, l1);
        float e0 = __expf(l0 - mx), e1 = __expf(l1 - mx), s = e0 + e1;
        out[doc * 2] = e0 / s; out[doc * 2 + 1] = e1 / s;
    }
}

// ---------------- launcher ----------------
extern "C" void kernel_launch(void* const* d_in, const int* in_sizes, int n_in,
                              void* d_out, int out_size, void* d_ws, size_t ws_size,
                              hipStream_t stream) {
    (void)in_sizes; (void)n_in; (void)out_size; (void)ws_size;
    const int*   x          = (const int*)d_in[0];
    const int*   text_masks = (const int*)d_in[1];
    const int*   post_masks = (const int*)d_in[2];
    const float* embed      = (const float*)d_in[3];
    const float* Wih_f      = (const float*)d_in[4];
    const float* Whh_f      = (const float*)d_in[5];
    const float* bih_f      = (const float*)d_in[6];
    const float* bhh_f      = (const float*)d_in[7];
    const float* Wih_b      = (const float*)d_in[8];
    const float* Whh_b      = (const float*)d_in[9];
    const float* bih_b      = (const float*)d_in[10];
    const float* bhh_b      = (const float*)d_in[11];
    const float* word_query = (const float*)d_in[12];
    const float* word_fc_W  = (const float*)d_in[13];
    const float* word_fc_b  = (const float*)d_in[14];
    const float* conv_W0    = (const float*)d_in[15];
    const float* conv_b0    = (const float*)d_in[16];
    const float* conv_W1    = (const float*)d_in[17];
    const float* conv_b1    = (const float*)d_in[18];
    const float* conv_W2    = (const float*)d_in[19];
    const float* conv_b2    = (const float*)d_in[20];
    const float* sent_query = (const float*)d_in[21];
    const float* sent_fc_W  = (const float*)d_in[22];
    const float* sent_fc_b  = (const float*)d_in[23];
    const float* mlp_W1     = (const float*)d_in[24];
    const float* mlp_b1     = (const float*)d_in[25];
    const float* mlp_W2     = (const float*)d_in[26];
    const float* mlp_b2     = (const float*)d_in[27];

    char* base = (char*)d_ws;
    size_t off = 0;
    auto alloc = [&](size_t bytes) -> void* {
        void* r = base + off;
        off = (off + bytes + 255) & ~(size_t)255;
        return r;
    };
    float* wo     = (float*)alloc((size_t)NSENT * SL * D2DIM * sizeof(float));
    bf16*  wo_bf  = (bf16*) alloc((size_t)NSENT * SL * D2DIM * sizeof(bf16));
    float* sv     = (float*)alloc((size_t)NSENT * CD3 * sizeof(float));
    float* pair   = (float*)alloc((size_t)NSENT * D2DIM * sizeof(float));
    float* sscore = (float*)alloc((size_t)NSENT * sizeof(float));
    bf16*  WihT_f = (bf16*)alloc((size_t)N3G * KXPAD * sizeof(bf16));
    bf16*  WhhT_f = (bf16*)alloc((size_t)N3G * GDIM * sizeof(bf16));
    bf16*  WihT_b = (bf16*)alloc((size_t)N3G * KXPAD * sizeof(bf16));
    bf16*  WhhT_b = (bf16*)alloc((size_t)N3G * GDIM * sizeof(bf16));
    bf16*  wfcT   = (bf16*)alloc((size_t)D2DIM * D2DIM * sizeof(bf16));
    bf16*  cT0    = (bf16*)alloc((size_t)112 * 512 * sizeof(bf16));
    bf16*  cT1    = (bf16*)alloc((size_t)112 * 768 * sizeof(bf16));
    bf16*  cT2    = (bf16*)alloc((size_t)112 * 1024 * sizeof(bf16));

    const int TH = 256;
    k_convert_pad<<<(N3G * KXPAD + TH - 1) / TH, TH, 0, stream>>>(Wih_f, WihT_f, N3G, EDIM, KXPAD);
    k_convert_pad<<<(N3G * GDIM  + TH - 1) / TH, TH, 0, stream>>>(Whh_f, WhhT_f, N3G, GDIM, GDIM);
    k_convert_pad<<<(N3G * KXPAD + TH - 1) / TH, TH, 0, stream>>>(Wih_b, WihT_b, N3G, EDIM, KXPAD);
    k_convert_pad<<<(N3G * GDIM  + TH - 1) / TH, TH, 0, stream>>>(Whh_b, WhhT_b, N3G, GDIM, GDIM);
    k_convert_pad<<<(D2DIM * D2DIM + TH - 1) / TH, TH, 0, stream>>>(word_fc_W, wfcT, D2DIM, D2DIM, D2DIM);
    k_conv_t<<<(112 * 512  + TH - 1) / TH, TH, 0, stream>>>(conv_W0, cT0, 512, 112);
    k_conv_t<<<(112 * 768  + TH - 1) / TH, TH, 0, stream>>>(conv_W1, cT1, 768, 112);
    k_conv_t<<<(112 * 1024 + TH - 1) / TH, TH, 0, stream>>>(conv_W2, cT2, 1024, 112);

    k_gru<<<dim3(NSENT / 16, 2), 256, 0, stream>>>(
        x, post_masks, embed, WihT_f, WhhT_f, bih_f, bhh_f,
        WihT_b, WhhT_b, bih_b, bhh_b, wo, wo_bf);

    k_sentence<<<NSENT, 256, 0, stream>>>(
        wo, wo_bf, post_masks, cT0, cT1, cT2, conv_b0, conv_b1, conv_b2,
        wfcT, word_fc_b, word_query, sv, pair);

    k_sentscore<<<NSENT, 256, 0, stream>>>(sv, sent_fc_W, sent_fc_b, sent_query, sscore);

    k_doc<<<BDOC, 256, 0, stream>>>(sv, pair, sscore, text_masks,
                                    mlp_W1, mlp_b1, mlp_W2, mlp_b2, (float*)d_out);
}
